// MambaMixer_32366873542667
// MI455X (gfx1250) — matrix operand order repaired and checked by reading, not compiled, not run on hardware
//
#include <hip/hip_runtime.h>
#include <hip/hip_bf16.h>
#include <stdint.h>

#define STR2(x) #x
#define STR(x) STR2(x)
#pragma message("CDNA5 build probe: __clang_major__=" STR(__clang_major__))

typedef __bf16 bf16_t;
typedef __attribute__((ext_vector_type(16))) __bf16 v16bf;
typedef __attribute__((ext_vector_type(8)))  float  v8f;
typedef __attribute__((ext_vector_type(4)))  unsigned int u32x4;
typedef __attribute__((ext_vector_type(4)))  int i32x4;
typedef __attribute__((ext_vector_type(8)))  int i32x8;

#if __has_builtin(__builtin_amdgcn_tensor_load_to_lds) && __has_builtin(__builtin_amdgcn_s_wait_tensorcnt)
#define HAVE_TDM 1
#pragma message("CDNA5 build probe: TDM builtin available")
#endif

// Problem sizes (fixed by the reference)
static const int kB     = 2;
static const int kL     = 1024;
static const int kDM    = 2048;          // D_MODEL
static const int kDI    = 4096;          // D_INNER
static const int kDS    = 16;            // D_STATE
static const int kDR    = 128;           // DT_RANK
static const int kRows  = kB * kL;       // 2048 token rows
static const int kNProj = 2 * kDI;       // 8192
static const int kNXd   = kDR + 2*kDS;   // 160

// ---------------------------------------------------------------------------
// helpers
// ---------------------------------------------------------------------------
__device__ inline bf16_t f2bf(float f) {
    unsigned x = __float_as_uint(f);
    unsigned r = (x + 0x7FFFu + ((x >> 16) & 1u)) >> 16;
    unsigned short s = (unsigned short)r;
    return *(bf16_t*)&s;
}

__global__ void f32_to_bf16_kernel(const float* __restrict__ in,
                                   bf16_t* __restrict__ out, size_t n) {
    size_t i = (size_t)blockIdx.x * 256 + threadIdx.x;
    if (i < n) out[i] = f2bf(in[i]);
}

// ---------------------------------------------------------------------------
// WMMA bf16 GEMM:  C[M,N] = A[M,K] * W[N,K]^T   (fp32 accumulate)
// Block tile 128x128, 256 threads = 8 waves, wave tile 64(M) x 32(N).
// A/B tiles (128x64 bf16) staged to LDS by the TDM, double buffered:
// 16 WMMAs per barrier pair.
// ---------------------------------------------------------------------------
#define TILE_M 128
#define TILE_N 128
#define TILE_K 64

// A fragment (16x32 bf16) at row m0, K offset k0 within the LDS tile.
__device__ inline v16bf load_a_frag(const bf16_t* lds, int m0, int k0, int lane) {
    union { v16bf v; unsigned u[8]; } f;
    int row   = m0 + (lane & 15);
    int khalf = (lane >> 4) * 8;
    const unsigned* base = (const unsigned*)(lds + row * TILE_K + k0);
#pragma unroll
    for (int v = 0; v < 8; ++v) {
        int k = ((v & 4) ? 16 : 0) + khalf + (v & 3) * 2;
        f.u[v] = base[k >> 1];
    }
    return f.v;
}

// B fragment (32x16 bf16) at col n0, K offset k0 within the LDS tile.
__device__ inline v16bf load_b_frag(const bf16_t* lds, int n0, int k0, int lane) {
    union { v16bf v; unsigned u[8]; } f;
    int col   = n0 + (lane & 15);
    int khalf = (lane >> 4) * 8;
    const unsigned* base = (const unsigned*)(lds + col * TILE_K + k0);
#pragma unroll
    for (int v = 0; v < 8; ++v) {
        int k = ((v & 4) ? 16 : 0) + khalf + (v & 3) * 2;
        f.u[v] = base[k >> 1];
    }
    return f.v;
}

#ifdef HAVE_TDM
// Issue a TDM 2D tile load: tileW x tileH bf16 elements from a row-major
// matrix (row stride strideElems) into LDS at ldsOff. remW/remH are the
// remaining tensor extents from the tile origin (TDM zero-fills OOB reads).
__device__ __forceinline__ void tdm_load_tile_bf16(
    unsigned ldsOff, const bf16_t* g, unsigned remW, unsigned remH,
    unsigned tileW, unsigned tileH, unsigned strideElems) {
    unsigned long long ga = (unsigned long long)(uintptr_t)g;
    u32x4 g0;
    g0[0] = 0x1u;                                   // count=1, no gather
    g0[1] = ldsOff;                                 // lds_addr (bytes)
    g0[2] = (unsigned)(ga & 0xFFFFFFFFu);           // global_addr[31:0]
    g0[3] = (unsigned)((ga >> 32) & 0x01FFFFFFu)    // global_addr[56:32]
          | (2u << 30);                             // type=2 (image)
    i32x8 g1;
    g1[0] = (int)(1u << 16);                        // data_size=1 (2 bytes)
    g1[1] = (int)((remW & 0xFFFFu) << 16);          // tensor_dim0[15:0]
    g1[2] = (int)((remW >> 16) | ((remH & 0xFFFFu) << 16));   // dim0 hi | dim1 lo
    g1[3] = (int)((remH >> 16) | (tileW << 16));    // dim1 hi | tile_dim0
    g1[4] = (int)(tileH & 0xFFFFu);                 // tile_dim1 (tile_dim2=0)
    g1[5] = (int)strideElems;                       // tensor_dim0_stride[31:0]
    g1[6] = 0;                                      // stride hi / dim1_stride lo
    g1[7] = 0;
    i32x4 z4 = {0, 0, 0, 0};
#if __clang_major__ >= 23
    i32x8 z8 = {0, 0, 0, 0, 0, 0, 0, 0};
    __builtin_amdgcn_tensor_load_to_lds(g0, g1, z4, z4, z8, 0);
#else
    __builtin_amdgcn_tensor_load_to_lds(g0, g1, z4, z4, 0);
#endif
}
#endif

__global__ __launch_bounds__(256)
void gemm_bf16_wmma(const bf16_t* __restrict__ A,   // [M,K] row-major
                    const bf16_t* __restrict__ W,   // [N,K] row-major (B^T)
                    float* __restrict__ C,          // [M,N] row-major
                    int M, int N, int K) {
    __shared__ __align__(16) bf16_t As[2][TILE_M * TILE_K];
    __shared__ __align__(16) bf16_t Bs[2][TILE_N * TILE_K];

    const int tid    = threadIdx.x;
    const int lane   = tid & 31;
    const int wave   = tid >> 5;
    const int mBlock = blockIdx.y * TILE_M;
    const int nBlock = blockIdx.x * TILE_N;
    const int mWave  = (wave & 1) * 64;   // 2 waves over M
    const int nWave  = (wave >> 1) * 32;  // 4 waves over N

    const v8f vzero = {0.f,0.f,0.f,0.f,0.f,0.f,0.f,0.f};
    v8f acc[4][2];
#pragma unroll
    for (int mi = 0; mi < 4; ++mi)
#pragma unroll
        for (int ni = 0; ni < 2; ++ni) acc[mi][ni] = vzero;

    const int nTiles = K / TILE_K;

#ifdef HAVE_TDM
    // -------- TDM double-buffered pipeline --------
    const unsigned remH_A = (unsigned)(M - mBlock);
    const unsigned remH_B = (unsigned)(N - nBlock);  // short tail -> zero fill

    if (wave == 0) {
        tdm_load_tile_bf16((unsigned)(uintptr_t)&As[0][0],
                           A + (size_t)mBlock * K, (unsigned)K, remH_A,
                           TILE_K, TILE_M, (unsigned)K);
        tdm_load_tile_bf16((unsigned)(uintptr_t)&Bs[0][0],
                           W + (size_t)nBlock * K, (unsigned)K, remH_B,
                           TILE_K, TILE_N, (unsigned)K);
    }

    for (int i = 0; i < nTiles; ++i) {
        const int buf = i & 1;
        if (wave == 0) {
            if (i + 1 < nTiles) {
                const int kt = (i + 1) * TILE_K;
                tdm_load_tile_bf16((unsigned)(uintptr_t)&As[buf ^ 1][0],
                                   A + (size_t)mBlock * K + kt,
                                   (unsigned)(K - kt), remH_A,
                                   TILE_K, TILE_M, (unsigned)K);
                tdm_load_tile_bf16((unsigned)(uintptr_t)&Bs[buf ^ 1][0],
                                   W + (size_t)nBlock * K + kt,
                                   (unsigned)(K - kt), remH_B,
                                   TILE_K, TILE_N, (unsigned)K);
                __builtin_amdgcn_s_wait_tensorcnt(2);  // tile i complete
            } else {
                __builtin_amdgcn_s_wait_tensorcnt(0);
            }
        }
        __syncthreads();

#pragma unroll
        for (int kk = 0; kk < TILE_K; kk += 32) {
            v16bf af[4], bfr[2];
#pragma unroll
            for (int mi = 0; mi < 4; ++mi)
                af[mi] = load_a_frag(As[buf], mWave + mi * 16, kk, lane);
#pragma unroll
            for (int ni = 0; ni < 2; ++ni)
                bfr[ni] = load_b_frag(Bs[buf], nWave + ni * 16, kk, lane);

#pragma unroll
            for (int mi = 0; mi < 4; ++mi)
#pragma unroll
                for (int ni = 0; ni < 2; ++ni)
                    acc[mi][ni] = __builtin_amdgcn_wmma_f32_16x16x32_bf16(
                        false, af[mi], false, bfr[ni],
                        (short)0, acc[mi][ni], false, false);
        }

        __syncthreads();  // protect buf before TDM refills it at i+2
    }
#else
    // -------- fallback: VGPR-staged tiles --------
    const int lr = tid >> 1;           // 0..127 rows
    const int lc = (tid & 1) * 32;     // 32 bf16 per thread
    const int nrow = nBlock + lr;
    for (int i = 0; i < nTiles; ++i) {
        const int kt = i * TILE_K;
        const uint4* ga = (const uint4*)(A + (size_t)(mBlock + lr) * K + kt + lc);
        uint4 a0 = ga[0], a1 = ga[1], a2 = ga[2], a3 = ga[3];
        uint4* sa = (uint4*)(&As[0][0] + lr * TILE_K + lc);
        sa[0] = a0; sa[1] = a1; sa[2] = a2; sa[3] = a3;
        uint4 b0 = make_uint4(0u,0u,0u,0u), b1 = b0, b2 = b0, b3 = b0;
        if (nrow < N) {
            const uint4* gb = (const uint4*)(W + (size_t)nrow * K + kt + lc);
            b0 = gb[0]; b1 = gb[1]; b2 = gb[2]; b3 = gb[3];
        }
        uint4* sb = (uint4*)(&Bs[0][0] + lr * TILE_K + lc);
        sb[0] = b0; sb[1] = b1; sb[2] = b2; sb[3] = b3;
        __syncthreads();
#pragma unroll
        for (int kk = 0; kk < TILE_K; kk += 32) {
            v16bf af[4], bfr[2];
#pragma unroll
            for (int mi = 0; mi < 4; ++mi)
                af[mi] = load_a_frag(&As[0][0], mWave + mi * 16, kk, lane);
#pragma unroll
            for (int ni = 0; ni < 2; ++ni)
                bfr[ni] = load_b_frag(&Bs[0][0], nWave + ni * 16, kk, lane);
#pragma unroll
            for (int mi = 0; mi < 4; ++mi)
#pragma unroll
                for (int ni = 0; ni < 2; ++ni)
                    acc[mi][ni] = __builtin_amdgcn_wmma_f32_16x16x32_bf16(
                        false, af[mi], false, bfr[ni],
                        (short)0, acc[mi][ni], false, false);
        }
        __syncthreads();
    }
#endif

    // store D: VGPR r -> M = r + (lane>=16 ? 8:0), N = lane&15
#pragma unroll
    for (int mi = 0; mi < 4; ++mi) {
#pragma unroll
        for (int ni = 0; ni < 2; ++ni) {
            int col = nBlock + nWave + ni * 16 + (lane & 15);
            if (col < N) {
                int rowBase = mBlock + mWave + mi * 16 + ((lane >> 4) << 3);
#pragma unroll
                for (int r = 0; r < 8; ++r)
                    C[(size_t)(rowBase + r) * N + col] = acc[mi][ni][r];
            }
        }
    }
}

// ---------------------------------------------------------------------------
// causal depthwise conv (d_conv=4) + SiLU:  u = silu(conv(xs))
// ---------------------------------------------------------------------------
__global__ __launch_bounds__(256)
void conv_silu_kernel(const float* __restrict__ proj,
                      const float* __restrict__ cw,  // [DI,4]
                      const float* __restrict__ cb,  // [DI]
                      float* __restrict__ u, bf16_t* __restrict__ ub) {
    size_t i = (size_t)blockIdx.x * 256 + threadIdx.x;  // over kRows*kDI
    int d = (int)(i % kDI);
    size_t row = i / kDI;            // b*L + l
    int l = (int)(row % kL);
    float acc = cb[d];
#pragma unroll
    for (int k = 0; k < 4; ++k) {
        int ls = l + k - 3;
        if (ls >= 0)
            acc += proj[(row + (size_t)(ls - l)) * kNProj + d] * cw[d * 4 + k];
    }
    float s = acc / (1.f + __expf(-acc));  // silu
    u[i]  = s;
    ub[i] = f2bf(s);
}

__global__ void slice_dtr_kernel(const float* __restrict__ xdbl,
                                 bf16_t* __restrict__ dtr) {
    int i = blockIdx.x * 256 + threadIdx.x;  // kRows*kDR
    int row = i / kDR, j = i % kDR;
    dtr[i] = f2bf(xdbl[row * kNXd + j]);
}

// ---------------------------------------------------------------------------
// selective scan, fused softplus + dA + SiLU(z) + D skip; one thread per (b,d)
// ---------------------------------------------------------------------------
__global__ __launch_bounds__(256)
void scan_kernel(const float* __restrict__ xdbl,   // [rows,160] (B at 128, C at 144)
                 const float* __restrict__ dtlin,  // [rows,DI] pre-softplus, no bias
                 const float* __restrict__ b_dt,   // [DI]
                 const float* __restrict__ A_log,  // [DI,16]
                 const float* __restrict__ u,      // [rows,DI]
                 const float* __restrict__ Dp,     // [DI]
                 const float* __restrict__ proj,   // [rows,8192] (z at +DI)
                 bf16_t* __restrict__ yb) {        // [rows,DI]
    __shared__ float BC[32];
    const int b = blockIdx.y;
    const int d = blockIdx.x * 256 + threadIdx.x;

    float As_[16], h[16];
#pragma unroll
    for (int s = 0; s < 16; ++s) {
        As_[s] = -__expf(A_log[d * 16 + s]);
        h[s] = 0.f;
    }
    const float Dd = Dp[d];
    const float bdt = b_dt[d];

    for (int t = 0; t < kL; ++t) {
        size_t row = (size_t)b * kL + t;
        if (threadIdx.x < 32)
            BC[threadIdx.x] = xdbl[row * kNXd + kDR + threadIdx.x];
        __syncthreads();

        float dl = dtlin[row * kDI + d] + bdt;
        float dt = (dl > 20.f) ? dl : log1pf(__expf(dl));
        float uv = u[row * kDI + d];
        float y = 0.f;
#pragma unroll
        for (int s = 0; s < 16; ++s) {
            float dA = __expf(dt * As_[s]);
            h[s] = dA * h[s] + dt * BC[s] * uv;
            y += h[s] * BC[16 + s];
        }
        float zv = proj[row * kNProj + kDI + d];
        float sz = zv / (1.f + __expf(-zv));  // silu(z)
        yb[row * kDI + d] = f2bf((y + uv * Dd) * sz);
        __syncthreads();
    }
}

// ---------------------------------------------------------------------------
// host launcher
// ---------------------------------------------------------------------------
static inline char* align256(char* p) {
    return (char*)(((uintptr_t)p + 255) & ~(uintptr_t)255);
}

extern "C" void kernel_launch(void* const* d_in, const int* in_sizes, int n_in,
                              void* d_out, int out_size, void* d_ws, size_t ws_size,
                              hipStream_t stream) {
    const float* x      = (const float*)d_in[0];  // [B,L,DM]
    const float* W_in   = (const float*)d_in[1];  // [8192,2048]
    const float* conv_w = (const float*)d_in[2];  // [4096,4]
    const float* conv_b = (const float*)d_in[3];  // [4096]
    const float* W_x    = (const float*)d_in[4];  // [160,4096]
    const float* W_dt   = (const float*)d_in[5];  // [4096,128]
    const float* b_dt   = (const float*)d_in[6];  // [4096]
    const float* A_log  = (const float*)d_in[7];  // [4096,16]
    const float* Dp     = (const float*)d_in[8];  // [4096]
    const float* W_out  = (const float*)d_in[9];  // [2048,4096]
    float* out = (float*)d_out;                   // [B,L,DM]

    char* p = (char*)d_ws;
    bf16_t* xb      = (bf16_t*)p; p = align256(p + sizeof(bf16_t)*(size_t)kRows*kDM);
    bf16_t* W_in_b  = (bf16_t*)p; p = align256(p + sizeof(bf16_t)*(size_t)kNProj*kDM);
    bf16_t* W_x_b   = (bf16_t*)p; p = align256(p + sizeof(bf16_t)*(size_t)kNXd*kDI);
    bf16_t* W_dt_b  = (bf16_t*)p; p = align256(p + sizeof(bf16_t)*(size_t)kDI*kDR);
    bf16_t* W_out_b = (bf16_t*)p; p = align256(p + sizeof(bf16_t)*(size_t)kDM*kDI);
    float*  proj    = (float*)p;  p = align256(p + sizeof(float)*(size_t)kRows*kNProj);
    float*  u       = (float*)p;  p = align256(p + sizeof(float)*(size_t)kRows*kDI);
    bf16_t* ub      = (bf16_t*)p; p = align256(p + sizeof(bf16_t)*(size_t)kRows*kDI);
    float*  xdbl    = (float*)p;  p = align256(p + sizeof(float)*(size_t)kRows*kNXd);
    bf16_t* dtr     = (bf16_t*)p; p = align256(p + sizeof(bf16_t)*(size_t)kRows*kDR);
    float*  dtlin   = (float*)p;  p = align256(p + sizeof(float)*(size_t)kRows*kDI);
    bf16_t* yb      = (bf16_t*)p; p = align256(p + sizeof(bf16_t)*(size_t)kRows*kDI);
    (void)ws_size; (void)in_sizes; (void)n_in; (void)out_size;

    auto cvt = [&](const float* src, bf16_t* dst, size_t n) {
        f32_to_bf16_kernel<<<dim3((unsigned)((n + 255) / 256)), 256, 0, stream>>>(src, dst, n);
    };
    cvt(x,     xb,      (size_t)kRows * kDM);
    cvt(W_in,  W_in_b,  (size_t)kNProj * kDM);
    cvt(W_x,   W_x_b,   (size_t)kNXd * kDI);
    cvt(W_dt,  W_dt_b,  (size_t)kDI * kDR);
    cvt(W_out, W_out_b, (size_t)kDM * kDI);

    auto gemm = [&](const bf16_t* A, const bf16_t* W, float* C, int M, int N, int K) {
        dim3 grid((N + TILE_N - 1) / TILE_N, (M + TILE_M - 1) / TILE_M);
        gemm_bf16_wmma<<<grid, 256, 0, stream>>>(A, W, C, M, N, K);
    };

    // 1) proj = x @ W_in^T   (2048 x 8192, K=2048)
    gemm(xb, W_in_b, proj, kRows, kNProj, kDM);

    // 2) u = silu(conv(xs))
    conv_silu_kernel<<<dim3((kRows * kDI) / 256), 256, 0, stream>>>(
        proj, conv_w, conv_b, u, ub);

    // 3) x_dbl = u @ W_x^T   (2048 x 160, K=4096)
    gemm(ub, W_x_b, xdbl, kRows, kNXd, kDI);

    // 4) dt_r -> bf16 ; dtlin = dt_r @ W_dt^T   (2048 x 4096, K=128)
    slice_dtr_kernel<<<dim3((kRows * kDR) / 256), 256, 0, stream>>>(xdbl, dtr);
    gemm(dtr, W_dt_b, dtlin, kRows, kDI, kDR);

    // 5) selective scan (fused softplus, dA, D-skip, silu(z)); writes yb (bf16)
    scan_kernel<<<dim3(kDI / 256, kB), 256, 0, stream>>>(
        xdbl, dtlin, b_dt, A_log, u, Dp, proj, yb);

    // 6) out = y @ W_out^T   (2048 x 2048, K=4096)
    gemm(yb, W_out_b, out, kRows, kDM, kDI);
}
